// MMDLoss_6820408066259
// MI455X (gfx1250) — matrix operand order repaired and checked by reading, not compiled, not run on hardware
//
#include <hip/hip_runtime.h>

typedef _Float16 v16h __attribute__((ext_vector_type(16)));
typedef _Float16 v8h  __attribute__((ext_vector_type(8)));
typedef float    v8f  __attribute__((ext_vector_type(8)));

#define NROW 8192
#define NDIM 64
#define NBW  10

__device__ __forceinline__ float wave_sum(float v) {
    #pragma unroll
    for (int off = 16; off > 0; off >>= 1) v += __shfl_xor(v, off, 32);
    return v;
}
__device__ __forceinline__ float wave_max(float v) {
    #pragma unroll
    for (int off = 16; off > 0; off >>= 1) v = fmaxf(v, __shfl_xor(v, off, 32));
    return v;
}

// One wave (32 lanes) per row: log_softmax over 64 elements, store f16 copy
// for WMMA operands plus fp32 squared row norm.
__global__ void __launch_bounds__(256)
mmd_logsoftmax_kernel(const float* __restrict__ src0, const float* __restrict__ src1,
                      _Float16* __restrict__ xh, _Float16* __restrict__ yh,
                      float* __restrict__ aa, float* __restrict__ bb) {
    const int wave = (blockIdx.x * blockDim.x + threadIdx.x) >> 5;
    const int lane = threadIdx.x & 31;
    const float* src; _Float16* dst; float* nrm; int r;
    if (wave < NROW) { src = src0; dst = xh; nrm = aa; r = wave; }
    else             { src = src1; dst = yh; nrm = bb; r = wave - NROW; }

    const float v0 = src[r * NDIM + lane];
    const float v1 = src[r * NDIM + lane + 32];
    const float m  = wave_max(fmaxf(v0, v1));
    const float s  = wave_sum(__expf(v0 - m) + __expf(v1 - m));
    const float ls = __logf(s);
    const float l0 = v0 - m - ls;
    const float l1 = v1 - m - ls;
    dst[r * NDIM + lane]      = (_Float16)l0;
    dst[r * NDIM + lane + 32] = (_Float16)l1;
    const float sq = wave_sum(l0 * l0 + l1 * l1);
    if (lane == 0) nrm[r] = sq;
}

__global__ void mmd_zero_kernel(float* p, int n) {
    const int i = blockIdx.x * blockDim.x + threadIdx.x;
    if (i < n) p[i] = 0.0f;
}

// One wave per 16x16 tile (grid-strided). Two v_wmma_f32_16x16x32_f16 cover
// K=64, then d = aa - 2ab + bb is folded through 10 bandwidths of v_exp_f32
// without ever touching memory with the distance matrix.
__global__ void __launch_bounds__(256)
mmd_pass_kernel(const _Float16* __restrict__ A, const _Float16* __restrict__ B,
                const float* __restrict__ an, const float* __restrict__ bn,
                float* __restrict__ partial /* [NBW] */) {
    __shared__ float red[NBW];
    if (threadIdx.x < NBW) red[threadIdx.x] = 0.0f;
    __syncthreads();

    const int lane   = threadIdx.x & 31;
    const int gwave  = blockIdx.x * (blockDim.x >> 5) + (threadIdx.x >> 5);
    const int nwaves = gridDim.x * (blockDim.x >> 5);
    const int tilesPerRow = NROW / 16;                 // 512
    const int nTiles = tilesPerRow * tilesPerRow;      // 262144

    // -coef_b * log2(e), coef_b = 1/(2 bw^2), bw_b = 128^(b/9) -> coef = 0.5*2^(-14b/9)
    float kneg[NBW];
    #pragma unroll
    for (int b = 0; b < NBW; ++b)
        kneg[b] = -1.4426950408889634f * 0.5f *
                  __builtin_amdgcn_exp2f(-14.0f * (float)b / 9.0f);

    float acc[NBW];
    #pragma unroll
    for (int b = 0; b < NBW; ++b) acc[b] = 0.0f;

    const int hi  = lane >> 4;   // half-wave select
    const int sub = lane & 15;

    for (int t = gwave; t < nTiles; t += nwaves) {
        const int tm = t >> 9;
        const int tn = t & (tilesPerRow - 1);

        // A fragment (16x32 f16): lanes 0-15 row=sub, K 0-15 (contiguous);
        // lanes 16-31 row=sub, K 16-31.
        const _Float16* ap = A + (size_t)(tm * 16 + sub) * NDIM + hi * 16;
        v8h a0lo = *(const v8h*)(ap);
        v8h a0hi = *(const v8h*)(ap + 8);
        v8h a1lo = *(const v8h*)(ap + 32);
        v8h a1hi = *(const v8h*)(ap + 40);
        v16h a0 = __builtin_shufflevector(a0lo, a0hi, 0,1,2,3,4,5,6,7,8,9,10,11,12,13,14,15);
        v16h a1 = __builtin_shufflevector(a1lo, a1hi, 0,1,2,3,4,5,6,7,8,9,10,11,12,13,14,15);

        // B fragment (32x16 f16): lanes 0-15 col=sub hold K0-15 (contiguous),
        // lanes 16-31 hold K16-31.
        const _Float16* bp = B + (size_t)(tn * 16 + sub) * NDIM + hi * 16;
        v16h b0 = *(const v16h*)(bp);
        v16h b1 = *(const v16h*)(bp + 32);

        v8f c = {};
        c = __builtin_amdgcn_wmma_f32_16x16x32_f16(false, a0, false, b0, (short)0, c, false, false);
        c = __builtin_amdgcn_wmma_f32_16x16x32_f16(false, a1, false, b1, (short)0, c, false, false);

        // D layout: lane -> n = sub; VGPR v -> m = v + 8*hi
        const float  bnv = bn[tn * 16 + sub];
        const float* anp = an + tm * 16 + hi * 8;
        float d[8];
        #pragma unroll
        for (int v = 0; v < 8; ++v)
            d[v] = anp[v] - 2.0f * c[v] + bnv;

        #pragma unroll
        for (int b = 0; b < NBW; ++b) {
            float s = 0.0f;
            #pragma unroll
            for (int v = 0; v < 8; ++v)
                s += __builtin_amdgcn_exp2f(d[v] * kneg[b]);
            acc[b] += s;
        }
    }

    #pragma unroll
    for (int b = 0; b < NBW; ++b) {
        const float s = wave_sum(acc[b]);
        if (lane == 0) atomicAdd(&red[b], s);
    }
    __syncthreads();
    if (threadIdx.x < NBW) atomicAdd(&partial[threadIdx.x], red[threadIdx.x]);
}

__global__ void mmd_finalize_kernel(const float* __restrict__ partial,
                                    float* __restrict__ out) {
    if (threadIdx.x == 0 && blockIdx.x == 0) {
        const float inv = 1.0f / ((float)NROW * (float)NROW);
        float m = 0.0f;
        for (int b = 0; b < NBW; ++b) {
            const float xx = partial[b];
            const float yy = partial[NBW + b];
            const float xy = partial[2 * NBW + b];
            m += (xx + yy - 2.0f * xy) * inv;
        }
        out[0] = m / (float)NBW;
    }
}

extern "C" void kernel_launch(void* const* d_in, const int* in_sizes, int n_in,
                              void* d_out, int out_size, void* d_ws, size_t ws_size,
                              hipStream_t stream) {
    (void)in_sizes; (void)n_in; (void)out_size; (void)ws_size;
    const float* src = (const float*)d_in[0];
    const float* tgt = (const float*)d_in[1];

    char* ws = (char*)d_ws;
    _Float16* xh = (_Float16*)(ws);                        // 1 MB
    _Float16* yh = (_Float16*)(ws + (1u << 20));           // 1 MB
    float* aa      = (float*)(ws + (2u << 20));            // 32 KB
    float* bb      = (float*)(ws + (2u << 20) + 32768);    // 32 KB
    float* partial = (float*)(ws + (2u << 20) + 65536);    // 30 floats

    mmd_zero_kernel<<<1, 32, 0, stream>>>(partial, 3 * NBW);
    // 16384 rows total, 8 waves per 256-thread block -> 2048 blocks
    mmd_logsoftmax_kernel<<<2048, 256, 0, stream>>>(src, tgt, xh, yh, aa, bb);
    // three fused GEMM+exp passes: xx, yy, xy
    mmd_pass_kernel<<<1024, 256, 0, stream>>>(xh, xh, aa, aa, partial + 0);
    mmd_pass_kernel<<<1024, 256, 0, stream>>>(yh, yh, bb, bb, partial + NBW);
    mmd_pass_kernel<<<1024, 256, 0, stream>>>(xh, yh, aa, bb, partial + 2 * NBW);
    mmd_finalize_kernel<<<1, 32, 0, stream>>>(partial, (float*)d_out);
}
